// DeformableTransformer_21543555957434
// MI455X (gfx1250) — matrix operand order repaired and checked by reading, not compiled, not run on hardware
//
#include <hip/hip_runtime.h>
#include <math.h>

// ---- problem constants (match reference) ----
#define D      256
#define NH     8
#define NL     4
#define NPNT   4
#define DFF    1024
#define BS     4
#define NQ     900
#define DH     32
#define S_TOT  21760
#define M0     (BS * NQ)      // 3600
#define PK     928            // NQ padded to multiple of 32 for attention K
#define E32    (BS * NH)      // 32 attention (batch,head) pairs
#define SCALE  0.17677669529663687f  // DH^-0.5

typedef unsigned short u16;
typedef unsigned int   u32;
typedef __attribute__((ext_vector_type(16))) __bf16   v16bf;
typedef __attribute__((ext_vector_type(8)))  float    v8f;
typedef __attribute__((ext_vector_type(8)))  unsigned v8u;

// pack two fp32 into bf16x2 (truncate) with one v_perm_b32
__device__ __forceinline__ unsigned pkbf(float lo, float hi) {
  return __builtin_amdgcn_perm(__builtin_bit_cast(unsigned, hi),
                               __builtin_bit_cast(unsigned, lo), 0x07060302u);
}
__device__ __forceinline__ u16 f2bf(float f) {
  return (u16)(__builtin_bit_cast(u32, f) >> 16);
}
// build a 16-element bf16 fragment from two 16B (8-elem) aligned runs
__device__ __forceinline__ v16bf frag16(const u16* p0, const u16* p1) {
  uint4 a = *(const uint4*)p0;
  uint4 b = *(const uint4*)p1;
  v8u u;
  u[0] = a.x; u[1] = a.y; u[2] = a.z; u[3] = a.w;
  u[4] = b.x; u[5] = b.y; u[6] = b.z; u[7] = b.w;
  return __builtin_bit_cast(v16bf, u);
}
__device__ __forceinline__ v8f wmma_bf16(v16bf a, v16bf b, v8f c) {
  return __builtin_amdgcn_wmma_f32_16x16x32_bf16(false, a, false, b, (short)0, c,
                                                 false, false);
}

#define GF_RELU 1
#define GF_C16  2   // store output as bf16 instead of fp32

// bf16 WMMA GEMM: C[M,N] = alpha * A[M,K] @ W[N,K]^T + bias + resid (+relu).
// A,B pre-staged in bf16; K multiple of 32 with pads zeroed at staging time.
// One wave per 32x32 tile (2x2 WMMA). Loads unconditional (indices clamped),
// divergence wave-uniform => EXEC all-ones at every WMMA.
__global__ __launch_bounds__(256) void gemm_bf16(
    const u16* __restrict__ A, int lda,
    const u16* __restrict__ B, int ldb,
    const float* __restrict__ bias,
    const float* __restrict__ resid, int ldr,
    void* __restrict__ Cp, int ldc,
    int M, int N, int K, float alpha, int flags)
{
  const int lane  = threadIdx.x & 31;
  const int wave  = threadIdx.x >> 5;
  const int nTn   = (N + 31) >> 5;
  const int nTm   = (M + 31) >> 5;
  const int tile  = blockIdx.x * 8 + wave;
  if (tile >= nTm * nTn) return;                 // wave-uniform exit
  const int tm    = tile / nTn;
  const int tn    = tile - tm * nTn;
  const int l15   = lane & 15;
  const int khalf = lane >> 4;
  const int ra0 = min(tm * 32 + l15,      M - 1);
  const int ra1 = min(tm * 32 + 16 + l15, M - 1);
  const int cb0 = min(tn * 32 + l15,      N - 1);
  const int cb1 = min(tn * 32 + 16 + l15, N - 1);
  const u16* a0 = A + (size_t)ra0 * lda;
  const u16* a1 = A + (size_t)ra1 * lda;
  const u16* b0 = B + (size_t)cb0 * ldb;
  const u16* b1 = B + (size_t)cb1 * ldb;

  v8f acc00 = {}, acc01 = {}, acc10 = {}, acc11 = {};
  for (int k0 = 0; k0 < K; k0 += 32) {
    const int kb0 = k0 + khalf * 8;   // A: elems 0..7  = K kb0..kb0+7
    const int kb1 = kb0 + 16;         //    elems 8..15 = K kb1..kb1+7
    const int kB  = k0 + khalf * 16;  // B: same K order as A (kb0, kb1)
    if (k0 + 32 < K) {
      __builtin_prefetch(a0 + k0 + 32, 0, 1);   // global_prefetch_b8
      __builtin_prefetch(b0 + k0 + 32, 0, 1);
    }
    const v16bf af0 = frag16(a0 + kb0, a0 + kb1);
    const v16bf af1 = frag16(a1 + kb0, a1 + kb1);
    const v16bf bf0 = frag16(b0 + kb0, b0 + kb1);
    const v16bf bf1 = frag16(b1 + kb0, b1 + kb1);
    acc00 = wmma_bf16(af0, bf0, acc00);
    acc01 = wmma_bf16(af0, bf1, acc01);
    acc10 = wmma_bf16(af1, bf0, acc10);
    acc11 = wmma_bf16(af1, bf1, acc11);
  }

  // epilogue: VGPR r -> lanes 0-15: M=r, lanes 16-31: M=r+8; N=lane&15
  float* C32 = (float*)Cp;
  u16*   C16 = (u16*)Cp;
  const int n0 = tn * 32 + l15;
  const int n1 = n0 + 16;
  const float bv0 = bias ? bias[min(n0, N - 1)] : 0.f;
  const float bv1 = bias ? bias[min(n1, N - 1)] : 0.f;
#pragma unroll
  for (int r = 0; r < 8; ++r) {
    const int m0 = tm * 32 + khalf * 8 + r;
    const int m1 = m0 + 16;
    const int mm[2] = {m0, m1};
    const int nn[2] = {n0, n1};
    const float bb[2] = {bv0, bv1};
    const float av[2][2] = {{acc00[r], acc01[r]}, {acc10[r], acc11[r]}};
#pragma unroll
    for (int i = 0; i < 2; ++i)
#pragma unroll
      for (int jn = 0; jn < 2; ++jn) {
        if (mm[i] < M && nn[jn] < N) {
          float v = av[i][jn] * alpha + bb[jn];
          if (resid) v += resid[(size_t)mm[i] * ldr + nn[jn]];
          if (flags & GF_RELU) v = fmaxf(v, 0.f);
          if (flags & GF_C16) C16[(size_t)mm[i] * ldc + nn[jn]] = f2bf(v);
          else                C32[(size_t)mm[i] * ldc + nn[jn]] = v;
        }
      }
  }
}

// fp32 -> bf16, 4 elements per thread (all sizes are multiples of 4)
__global__ void cvt_bf16_k(const float* __restrict__ x, u16* __restrict__ y, int n) {
  int i = (blockIdx.x * 256 + threadIdx.x) * 4;
  if (i + 4 <= n) {
    float4 f = *(const float4*)(x + i);
    uint2 o; o.x = pkbf(f.x, f.y); o.y = pkbf(f.z, f.w);
    *(uint2*)(y + i) = o;
  }
}

// y_bf16 = a + b (fused residual-add + convert)
__global__ void add_cvt_k(const float* __restrict__ a, const float* __restrict__ b,
                          u16* __restrict__ y, int n) {
  int i = (blockIdx.x * 256 + threadIdx.x) * 4;
  if (i + 4 <= n) {
    float4 fa = *(const float4*)(a + i);
    float4 fb = *(const float4*)(b + i);
    uint2 o; o.x = pkbf(fa.x + fb.x, fa.y + fb.y); o.y = pkbf(fa.z + fb.z, fa.w + fb.w);
    *(uint2*)(y + i) = o;
  }
}

// one wave per attention row: softmax over 900 fp32 scores -> bf16 probs,
// zeroing the 28 pad columns so the padded-K A*V GEMM is exact
__global__ __launch_bounds__(256) void softmax_bf16(
    const float* __restrict__ P, u16* __restrict__ Pb, int rows)
{
  const int lane = threadIdx.x & 31;
  const int r = blockIdx.x * 8 + (threadIdx.x >> 5);
  if (r >= rows) return;
  const float* row = P + (size_t)r * PK;
  u16* orow = Pb + (size_t)r * PK;
  float m = -3.4e38f;
  for (int i = lane; i < NQ; i += 32) m = fmaxf(m, row[i]);
  for (int off = 16; off; off >>= 1) m = fmaxf(m, __shfl_xor(m, off));
  float s = 0.f;
  for (int i = lane; i < NQ; i += 32) s += __expf(row[i] - m);
  for (int off = 16; off; off >>= 1) s += __shfl_xor(s, off);
  const float inv = 1.f / s;
  for (int i = lane; i < NQ; i += 32) orow[i] = f2bf(__expf(row[i] - m) * inv);
  for (int i = NQ + lane; i < PK; i += 32) orow[i] = 0;
}

// vT[e][n][k] = vp16[b*NQ+k][h*DH+n], zero-padded to k<PK  (e = b*NH+h)
__global__ void transpose_v(const u16* __restrict__ vp16, u16* __restrict__ vT) {
  const int total = E32 * DH * PK;
  int idx = blockIdx.x * 256 + threadIdx.x;
  if (idx >= total) return;
  int k = idx % PK;
  int t = idx / PK;
  int n = t % DH;
  int e = t / DH;
  int b = e >> 3, h = e & 7;
  u16 v = 0;
  if (k < NQ) v = vp16[(size_t)(b * NQ + k) * D + h * DH + n];
  vT[idx] = v;
}

// out32 = LN(x) * g + b ; optional extra bf16 copy (row length D=256)
__global__ __launch_bounds__(256) void layernorm_k(
    const float* __restrict__ x,
    const float* __restrict__ g, const float* __restrict__ b,
    float* __restrict__ out32, u16* __restrict__ out16)
{
  __shared__ float red[256];
  const int row = blockIdx.x;
  const int t = threadIdx.x;
  float v = x[(size_t)row * D + t];
  red[t] = v; __syncthreads();
  for (int s = 128; s; s >>= 1) { if (t < s) red[t] += red[t + s]; __syncthreads(); }
  const float mean = red[0] * (1.f / D);
  __syncthreads();
  const float dv = v - mean;
  red[t] = dv * dv; __syncthreads();
  for (int s = 128; s; s >>= 1) { if (t < s) red[t] += red[t + s]; __syncthreads(); }
  const float var = red[0] * (1.f / D);
  const float o = dv * rsqrtf(var + 1e-5f) * g[t] + b[t];
  out32[(size_t)row * D + t] = o;
  if (out16) out16[(size_t)row * D + t] = f2bf(o);
}

// multi-scale deformable sampling: one wave per (b,q,h); lane == channel (DH==32)
__global__ __launch_bounds__(256) void msdeform_sample(
    const float* __restrict__ value,   // (BS, S, D) fp32
    const float* __restrict__ samp,    // (M0, NH*NL*NP*2)
    const float* __restrict__ awl,     // (M0, NH*NL*NP) logits
    const float* __restrict__ refp,    // (BS, NQ, NL, 4)
    u16* __restrict__ out16)           // (M0, D) bf16 (feeds next GEMM)
{
  const int lane = threadIdx.x & 31;
  const int w = blockIdx.x * 8 + (threadIdx.x >> 5);
  if (w >= M0 * NH) return;
  const int h  = w & (NH - 1);
  const int bq = w >> 3;
  const int b  = bq / NQ;
  const int HLs[4] = {128, 64, 32, 16};
  const int WLs[4] = {128, 64, 32, 16};
  const int S0s[4] = {0, 16384, 20480, 21504};

  const float* lg = awl + (size_t)bq * (NH * NL * NPNT) + h * 16;
  float mx = -3.4e38f;
#pragma unroll
  for (int j = 0; j < 16; ++j) mx = fmaxf(mx, lg[j]);
  float wj[16]; float ssum = 0.f;
#pragma unroll
  for (int j = 0; j < 16; ++j) { wj[j] = __expf(lg[j] - mx); ssum += wj[j]; }
  const float inv = 1.f / ssum;

  const float* sp    = samp + (size_t)bq * (NH * NL * NPNT * 2) + h * 32;
  const float* rp    = refp + (size_t)bq * (NL * 4);
  const float* vbase = value + (size_t)b * S_TOT * D + h * DH + lane;

  float acc = 0.f;
  for (int l = 0; l < NL; ++l) {
    const int Hl = HLs[l], Wl = WLs[l];
    const float cx = rp[l * 4 + 0], cy = rp[l * 4 + 1];
    const float cw = rp[l * 4 + 2], ch = rp[l * 4 + 3];
    const float* vl = vbase + (size_t)S0s[l] * D;
#pragma unroll
    for (int p = 0; p < NPNT; ++p) {
      const int j = l * NPNT + p;
      const float aw = wj[j] * inv;
      const float locx = cx + sp[j * 2 + 0] * (0.5f / NPNT) * cw;
      const float locy = cy + sp[j * 2 + 1] * (0.5f / NPNT) * ch;
      const float x = locx * Wl - 0.5f;
      const float y = locy * Hl - 0.5f;
      const float xf = floorf(x), yf = floorf(y);
      const float fx = x - xf, fy = y - yf;
      const int x0 = (int)xf, y0 = (int)yf;
      float s4 = 0.f;
#pragma unroll
      for (int c = 0; c < 4; ++c) {
        const int xi = x0 + (c & 1);
        const int yi = y0 + (c >> 1);
        const float wc = ((c & 1) ? fx : 1.f - fx) * ((c >> 1) ? fy : 1.f - fy);
        if (xi >= 0 && xi < Wl && yi >= 0 && yi < Hl)
          s4 += wc * vl[(size_t)(yi * Wl + xi) * D];
      }
      acc += aw * s4;
    }
  }
  out16[(size_t)bq * D + h * DH + lane] = f2bf(acc);
}

extern "C" void kernel_launch(void* const* d_in, const int* in_sizes, int n_in,
                              void* d_out, int out_size, void* d_ws, size_t ws_size,
                              hipStream_t stream)
{
  (void)in_sizes; (void)n_in; (void)out_size; (void)ws_size;
  const float* tgt    = (const float*)d_in[0];
  const float* qpos   = (const float*)d_in[1];
  const float* refp   = (const float*)d_in[2];
  const float* mem    = (const float*)d_in[3];
  const float* in_w   = (const float*)d_in[5];
  const float* in_b   = (const float*)d_in[6];
  const float* out_w  = (const float*)d_in[7];
  const float* out_b  = (const float*)d_in[8];
  const float* samp_w = (const float*)d_in[9];
  const float* samp_b = (const float*)d_in[10];
  const float* aw_w   = (const float*)d_in[11];
  const float* aw_b   = (const float*)d_in[12];
  const float* val_w  = (const float*)d_in[13];
  const float* val_b  = (const float*)d_in[14];
  const float* out2_w = (const float*)d_in[15];
  const float* out2_b = (const float*)d_in[16];
  const float* ln1_g  = (const float*)d_in[17];
  const float* ln1_b  = (const float*)d_in[18];
  const float* ln2_g  = (const float*)d_in[19];
  const float* ln2_b  = (const float*)d_in[20];
  const float* ln3_g  = (const float*)d_in[21];
  const float* ln3_b  = (const float*)d_in[22];
  const float* lin1_w = (const float*)d_in[23];
  const float* lin1_b = (const float*)d_in[24];
  const float* lin2_w = (const float*)d_in[25];
  const float* lin2_b = (const float*)d_in[26];

  char* base = (char*)d_ws;
  size_t off = 0;
  auto allocf = [&](size_t n) { float* p = (float*)(base + off); off = (off + n * 4 + 255) & ~(size_t)255; return p; };
  auto alloch = [&](size_t n) { u16*   p = (u16*)(base + off);  off = (off + n * 2 + 255) & ~(size_t)255; return p; };

  float* P    = allocf((size_t)E32 * NQ * PK);   // fp32 attention scores
  float* val  = allocf((size_t)BS * S_TOT * D);  // fp32 value (gathered later)
  float* smp  = allocf((size_t)M0 * 256);
  float* awl  = allocf((size_t)M0 * 128);
  float* proj = allocf((size_t)M0 * D);
  float* tgtA = allocf((size_t)M0 * D);
  float* tgtB = allocf((size_t)M0 * D);
  u16* q16    = alloch((size_t)M0 * D);
  u16* tgt16  = alloch((size_t)M0 * D);
  u16* qp16   = alloch((size_t)M0 * D);
  u16* kp16   = alloch((size_t)M0 * D);
  u16* vp16   = alloch((size_t)M0 * D);
  u16* attn16 = alloch((size_t)M0 * D);
  u16* mso16  = alloch((size_t)M0 * D);
  u16* tgtB16 = alloch((size_t)M0 * D);
  u16* ffn116 = alloch((size_t)M0 * DFF);
  u16* Pb     = alloch((size_t)E32 * NQ * PK);   // bf16 attention probs
  u16* vT     = alloch((size_t)E32 * DH * PK);   // bf16 V^T, zero-padded
  u16* mem16  = alloch((size_t)BS * S_TOT * D);
  u16* w_in   = alloch((size_t)3 * D * D);
  u16* w_out  = alloch((size_t)D * D);
  u16* w_samp = alloch((size_t)256 * D);
  u16* w_aw   = alloch((size_t)128 * D);
  u16* w_val  = alloch((size_t)D * D);
  u16* w_out2 = alloch((size_t)D * D);
  u16* w_lin1 = alloch((size_t)DFF * D);
  u16* w_lin2 = alloch((size_t)D * DFF);

  auto cvt = [&](const float* x, u16* y, size_t n) {
    cvt_bf16_k<<<(unsigned)((n / 4 + 255) / 256), 256, 0, stream>>>(x, y, (int)n);
  };
  auto gemm = [&](const u16* A, int lda, const u16* B, int ldb,
                  const float* bias, const float* resid, int ldr,
                  void* C, int ldc, int M, int N, int K, float alpha, int flags) {
    int tiles  = ((M + 31) / 32) * ((N + 31) / 32);
    int blocks = (tiles + 7) / 8;
    gemm_bf16<<<blocks, 256, 0, stream>>>(A, lda, B, ldb, bias, resid, ldr,
                                          C, ldc, M, N, K, alpha, flags);
  };

  // ---- stage weights + static activations to bf16 (one pass each) ----
  cvt(in_w,   w_in,   (size_t)3 * D * D);
  cvt(out_w,  w_out,  (size_t)D * D);
  cvt(samp_w, w_samp, (size_t)256 * D);
  cvt(aw_w,   w_aw,   (size_t)128 * D);
  cvt(val_w,  w_val,  (size_t)D * D);
  cvt(out2_w, w_out2, (size_t)D * D);
  cvt(lin1_w, w_lin1, (size_t)DFF * D);
  cvt(lin2_w, w_lin2, (size_t)D * DFF);
  cvt(tgt,    tgt16,  (size_t)M0 * D);
  cvt(mem,    mem16,  (size_t)BS * S_TOT * D);

  // ---- MHA self-attention ----
  add_cvt_k<<<(M0 * D / 4 + 255) / 256, 256, 0, stream>>>(tgt, qpos, q16, M0 * D);
  gemm(q16,   D, w_in,             D, in_b,         nullptr, 0, qp16, D, M0, D, D, 1.f, GF_C16);
  gemm(q16,   D, w_in + D * D,     D, in_b + D,     nullptr, 0, kp16, D, M0, D, D, 1.f, GF_C16);
  gemm(tgt16, D, w_in + 2 * D * D, D, in_b + 2 * D, nullptr, 0, vp16, D, M0, D, D, 1.f, GF_C16);
  for (int e = 0; e < E32; ++e) {
    const int b = e >> 3, h = e & 7;
    gemm(qp16 + (size_t)b * NQ * D + h * DH, D,
         kp16 + (size_t)b * NQ * D + h * DH, D,
         nullptr, nullptr, 0,
         P + (size_t)e * NQ * PK, PK, NQ, NQ, DH, SCALE, 0);
  }
  softmax_bf16<<<(E32 * NQ + 7) / 8, 256, 0, stream>>>(P, Pb, E32 * NQ);
  transpose_v<<<(E32 * DH * PK + 255) / 256, 256, 0, stream>>>(vp16, vT);
  for (int e = 0; e < E32; ++e) {
    const int b = e >> 3, h = e & 7;
    gemm(Pb + (size_t)e * NQ * PK, PK,
         vT + (size_t)e * DH * PK, PK,
         nullptr, nullptr, 0,
         attn16 + (size_t)b * NQ * D + h * DH, D, NQ, DH, PK, 1.f, GF_C16);
  }
  gemm(attn16, D, w_out, D, out_b, tgt, D, proj, D, M0, D, D, 1.f, 0);
  layernorm_k<<<M0, 256, 0, stream>>>(proj, ln2_g, ln2_b, tgtA, nullptr);

  // ---- multi-scale deformable attention ----
  add_cvt_k<<<(M0 * D / 4 + 255) / 256, 256, 0, stream>>>(tgtA, qpos, q16, M0 * D);
  gemm(mem16, D, w_val,  D, val_b,  nullptr, 0, val, D,   BS * S_TOT, D,   D, 1.f, 0);
  gemm(q16,   D, w_samp, D, samp_b, nullptr, 0, smp, 256, M0,         256, D, 1.f, 0);
  gemm(q16,   D, w_aw,   D, aw_b,   nullptr, 0, awl, 128, M0,         128, D, 1.f, 0);
  msdeform_sample<<<(M0 * NH + 7) / 8, 256, 0, stream>>>(val, smp, awl, refp, mso16);
  gemm(mso16, D, w_out2, D, out2_b, tgtA, D, proj, D, M0, D, D, 1.f, 0);
  layernorm_k<<<M0, 256, 0, stream>>>(proj, ln1_g, ln1_b, tgtB, tgtB16);

  // ---- FFN ----
  gemm(tgtB16, D,   w_lin1, D,   lin1_b, nullptr, 0, ffn116, DFF, M0, DFF, D,   1.f, GF_RELU | GF_C16);
  gemm(ffn116, DFF, w_lin2, DFF, lin2_b, tgtB,    D, proj,   D,   M0, D,   DFF, 1.f, 0);
  layernorm_k<<<M0, 256, 0, stream>>>(proj, ln3_g, ln3_b, (float*)d_out, nullptr);
}
